// GatingNetwork_77378130804781
// MI455X (gfx1250) — matrix operand order repaired and checked by reading, not compiled, not run on hardware
//
#include <hip/hip_runtime.h>
#include <hip/hip_bf16.h>

typedef __attribute__((ext_vector_type(16))) __bf16 v16bf;
typedef __attribute__((ext_vector_type(8)))  float  v8f;

#define TEMPERATURE 0.5f
#define DDIM  4096
#define EEXP  64
#define KTILES (DDIM / 32)   // 128 K-tiles of 32
#define ETILES (EEXP / 16)   // 4 expert tiles of 16

// ---------------------------------------------------------------------------
// Kernel 1: convert W [64 x 4096] f32 -> bf16, pre-swizzled into B-fragment
// order for v_wmma_f32_16x16x32_bf16.
// B (32x16, 16-bit) layout used here (contiguous K, same order as A): lane L holds column N = L%16;
// lanes 0-15 hold K=0..15, lanes 16-31 hold K=16..31, 2 values per VGPR.
// Storage: frag index f = kt*4 + et; element address:
//   ws[(f*32 + lane)*16 + i]  with  e = et*16 + lane%16,  K = kt*32 + (lane>>4)*16 + i
// => each lane's 16 bf16 (32B) are contiguous -> single vector load in GEMM.
// ---------------------------------------------------------------------------
__global__ void swizzle_W_bf16(const float* __restrict__ W,
                               unsigned short* __restrict__ ws) {
    int idx = blockIdx.x * blockDim.x + threadIdx.x;      // one bf16 element
    if (idx >= EEXP * DDIM) return;
    int i    = idx & 15;
    int lane = (idx >> 4) & 31;
    int et   = (idx >> 9) & 3;
    int kt   = idx >> 11;
    int e = et * 16 + (lane & 15);
    int k = kt * 32 + ((lane >> 4) << 4) + i;
    reinterpret_cast<__bf16*>(ws)[idx] = (__bf16)W[(size_t)e * DDIM + k];
}

__device__ __forceinline__ v16bf cvt_afrag(float4 a0, float4 a1, float4 a2, float4 a3) {
    v16bf af;
    af[0]  = (__bf16)a0.x; af[1]  = (__bf16)a0.y;
    af[2]  = (__bf16)a0.z; af[3]  = (__bf16)a0.w;
    af[4]  = (__bf16)a1.x; af[5]  = (__bf16)a1.y;
    af[6]  = (__bf16)a1.z; af[7]  = (__bf16)a1.w;
    af[8]  = (__bf16)a2.x; af[9]  = (__bf16)a2.y;
    af[10] = (__bf16)a2.z; af[11] = (__bf16)a2.w;
    af[12] = (__bf16)a3.x; af[13] = (__bf16)a3.y;
    af[14] = (__bf16)a3.z; af[15] = (__bf16)a3.w;
    return af;
}

// ---------------------------------------------------------------------------
// Kernel 2: GEMM (bf16 WMMA, f32 accumulate) + fused bias/scale + top-8
// masked-softmax epilogue. 128 threads = 4 wave32; each wave owns 16 rows.
// Software-pipelined: iteration kt+1's A/B loads issue before kt's wmmas.
// ---------------------------------------------------------------------------
__global__ __launch_bounds__(128)
void gating_gemm_topk(const float* __restrict__ g,
                      const unsigned short* __restrict__ Wb,
                      const float* __restrict__ bias,
                      const float* __restrict__ alpha,
                      const float* __restrict__ ebias,
                      float* __restrict__ out) {
    __shared__ float lds_logits[64 * 65];   // 64 rows, stride 65 (bank-conflict pad)

    const int tid    = threadIdx.x;
    const int lane   = tid & 31;
    const int wave   = tid >> 5;
    const int laneLo = lane & 15;
    const int laneHi = lane >> 4;

    const int rowBlock = blockIdx.x * 64;          // 64 rows per block
    const int rowWave  = rowBlock + wave * 16;     // 16 rows per wave

    v8f acc[ETILES] = {};

    // A-fragment source: lane holds row M = laneLo; K chunks match B fragment:
    // lanes 0-15 -> K = 0..15; lanes 16-31 -> K = 16..31.
    const float* arow = g + (size_t)(rowWave + laneLo) * DDIM + laneHi * 16;
    const v16bf* bfr  = reinterpret_cast<const v16bf*>(Wb);

    // ---- software pipeline: preload kt = 0 ----
    float4 a0, a1, a2, a3;
    v16bf bf0, bf1, bf2, bf3;
    {
        const float4* ap = (const float4*)arow;
        a0 = ap[0]; a1 = ap[1]; a2 = ap[2]; a3 = ap[3];
        const v16bf* bp = bfr + lane;
        bf0 = bp[0]; bf1 = bp[32]; bf2 = bp[64]; bf3 = bp[96];
    }

    for (int kt = 0; kt < KTILES - 1; ++kt) {
        // consume current buffers
        v16bf af = cvt_afrag(a0, a1, a2, a3);
        v16bf c0 = bf0, c1 = bf1, c2 = bf2, c3 = bf3;

        // issue next iteration's loads BEFORE the wmmas (latency hiding)
        const float4* ap = (const float4*)(arow + (kt + 1) * 32);
        a0 = ap[0]; a1 = ap[1]; a2 = ap[2]; a3 = ap[3];
        const v16bf* bp = bfr + (size_t)(kt + 1) * (ETILES * 32) + lane;
        bf0 = bp[0]; bf1 = bp[32]; bf2 = bp[64]; bf3 = bp[96];

        // stream-prefetch the A row 4KB (32 K-tiles) ahead -> global_prefetch_b8
        __builtin_prefetch(arow + (kt + 1) * 32 + 1024, 0, 1);

        acc[0] = __builtin_amdgcn_wmma_f32_16x16x32_bf16(false, af, false, c0,
                                                         (short)0, acc[0], false, false);
        acc[1] = __builtin_amdgcn_wmma_f32_16x16x32_bf16(false, af, false, c1,
                                                         (short)0, acc[1], false, false);
        acc[2] = __builtin_amdgcn_wmma_f32_16x16x32_bf16(false, af, false, c2,
                                                         (short)0, acc[2], false, false);
        acc[3] = __builtin_amdgcn_wmma_f32_16x16x32_bf16(false, af, false, c3,
                                                         (short)0, acc[3], false, false);
    }
    {   // peeled final iteration
        v16bf af = cvt_afrag(a0, a1, a2, a3);
        acc[0] = __builtin_amdgcn_wmma_f32_16x16x32_bf16(false, af, false, bf0,
                                                         (short)0, acc[0], false, false);
        acc[1] = __builtin_amdgcn_wmma_f32_16x16x32_bf16(false, af, false, bf1,
                                                         (short)0, acc[1], false, false);
        acc[2] = __builtin_amdgcn_wmma_f32_16x16x32_bf16(false, af, false, bf2,
                                                         (short)0, acc[2], false, false);
        acc[3] = __builtin_amdgcn_wmma_f32_16x16x32_bf16(false, af, false, bf3,
                                                         (short)0, acc[3], false, false);
    }

    // Scatter C fragments into LDS with the full logit transform applied:
    //   logit = (dot + b[e]) * alpha / T + expert_bias[e]
    // C/D layout: VGPR j, lanes 0-15 -> M=j, N=laneLo; lanes 16-31 -> M=j+8.
    const float scale = alpha[0] * (1.0f / TEMPERATURE);
#pragma unroll
    for (int et = 0; et < ETILES; ++et) {
        int c    = et * 16 + laneLo;
        float bc = bias[c];
        float ec = ebias[c];
#pragma unroll
        for (int j = 0; j < 8; ++j) {
            int r = wave * 16 + laneHi * 8 + j;
            lds_logits[r * 65 + c] = (acc[et][j] + bc) * scale + ec;
        }
    }
    __syncthreads();

    // Epilogue: one thread per row. 8 max-sweeps find the 8th-largest value
    // (the top-k threshold); then masked softmax renormalized over the top-8.
    if (tid < 64) {
        const float* lrow = lds_logits + tid * 65;
        float t  = 3.0e38f;
        float mx = -3.0e38f;
#pragma unroll 1
        for (int k = 0; k < 8; ++k) {
            float m = -3.0e38f;
            for (int e = 0; e < EEXP; ++e) {
                float v = lrow[e];
                if (v < t && v > m) m = v;
            }
            t = m;
            if (k == 0) mx = m;       // first sweep max == row max
        }
        float s = 0.0f;
        for (int e = 0; e < EEXP; ++e) {
            float v = lrow[e];
            if (v >= t) s += __expf(v - mx);
        }
        float inv = 1.0f / (s + 1e-12f);
        float4* orow = (float4*)(out + (size_t)(rowBlock + tid) * EEXP);
#pragma unroll 1
        for (int e4 = 0; e4 < EEXP / 4; ++e4) {
            float4 w;
            float v0 = lrow[e4 * 4 + 0];
            float v1 = lrow[e4 * 4 + 1];
            float v2 = lrow[e4 * 4 + 2];
            float v3 = lrow[e4 * 4 + 3];
            w.x = (v0 >= t) ? __expf(v0 - mx) * inv : 0.0f;
            w.y = (v1 >= t) ? __expf(v1 - mx) * inv : 0.0f;
            w.z = (v2 >= t) ? __expf(v2 - mx) * inv : 0.0f;
            w.w = (v3 >= t) ? __expf(v3 - mx) * inv : 0.0f;
            orow[e4] = w;
        }
    }
}

extern "C" void kernel_launch(void* const* d_in, const int* in_sizes, int n_in,
                              void* d_out, int out_size, void* d_ws, size_t ws_size,
                              hipStream_t stream) {
    (void)in_sizes; (void)n_in; (void)out_size; (void)ws_size;
    const float* g_emb = (const float*)d_in[0];
    const float* W     = (const float*)d_in[1];
    const float* b     = (const float*)d_in[2];
    const float* alpha = (const float*)d_in[3];
    const float* ebias = (const float*)d_in[4];
    unsigned short* Wb = (unsigned short*)d_ws;   // 64*4096*2 = 512 KB scratch
    float* out = (float*)d_out;

    // Kernel 1: W f32 -> bf16, pre-swizzled into B-fragment order.
    swizzle_W_bf16<<<(EEXP * DDIM + 255) / 256, 256, 0, stream>>>(W, Wb);

    // Kernel 2: 16384 rows / 64 rows-per-block = 256 blocks of 128 threads.
    gating_gemm_topk<<<16384 / 64, 128, 0, stream>>>(g_emb, Wb, b, alpha, ebias, out);
}
